// BEVTransformerBlock_41601053229133
// MI455X (gfx1250) — matrix operand order repaired and checked by reading, not compiled, not run on hardware
//
#include <hip/hip_runtime.h>
#include <hip/hip_bf16.h>
#include <math.h>

typedef __attribute__((ext_vector_type(16))) _Float16 v16h;
typedef __attribute__((ext_vector_type(8)))  _Float16 v8h;
typedef __attribute__((ext_vector_type(4)))  _Float16 v4h;
typedef __attribute__((ext_vector_type(8)))  float    v8f;
typedef __attribute__((ext_vector_type(4)))  int      i32x4;
typedef __attribute__((ext_vector_type(2)))  int      i32x2;

#define WIN   25
#define NTOK  625
#define NPAD  640
#define NWIN  32
#define DMODEL 256
#define NHEADS 8
#define DHEAD  32
#define HID    512
#define MROWS  (NWIN * NPAD)     // 20480
#define QSCALE 0.17677669529663687f

#if __has_builtin(__builtin_amdgcn_global_load_async_to_lds_b128)
#define HAVE_ASYNC128 1
#else
#define HAVE_ASYNC128 0
#endif
#if __has_builtin(__builtin_amdgcn_global_load_async_to_lds_b64)
#define HAVE_ASYNC64 1
#else
#define HAVE_ASYNC64 0
#endif

static __device__ __forceinline__ void async_b128(const void* g, void* l) {
#if HAVE_ASYNC128
  __builtin_amdgcn_global_load_async_to_lds_b128(
      (__attribute__((address_space(1))) i32x4*)g,
      (__attribute__((address_space(3))) i32x4*)l, 0, 0);
#else
  (void)g; (void)l;
#endif
}
static __device__ __forceinline__ void async_b64(const void* g, void* l) {
#if HAVE_ASYNC64
  __builtin_amdgcn_global_load_async_to_lds_b64(
      (__attribute__((address_space(1))) i32x2*)g,
      (__attribute__((address_space(3))) i32x2*)l, 0, 0);
#else
  (void)g; (void)l;
#endif
}
static __device__ __forceinline__ void wait_async() {
#if HAVE_ASYNC128 || HAVE_ASYNC64
#if __has_builtin(__builtin_amdgcn_s_wait_asynccnt)
  __builtin_amdgcn_s_wait_asynccnt(0);
#else
  asm volatile("s_wait_asynccnt 0" ::: "memory");
#endif
#endif
}

static __device__ __forceinline__ v8f wmma_f16(v16h a, v16h b, v8f c) {
  return __builtin_amdgcn_wmma_f32_16x16x32_f16(false, a, false, b, (short)0, c, false, false);
}

// ---------------- window partition: x (2,256,100,100) -> xw f16 (32,640,256), zero-padded rows
__global__ __launch_bounds__(256) void k_winpart(const float* __restrict__ x, _Float16* __restrict__ xw) {
  int idx = blockIdx.x * 256 + threadIdx.x;          // MROWS*DMODEL
  if (idx >= MROWS * DMODEL) return;
  int d = idx & 255;
  int row = idx >> 8;
  int token = row % NPAD;
  int win = row / NPAD;
  _Float16 v = (_Float16)0.f;
  if (token < NTOK) {
    int b = win >> 4, wi = win & 15;
    int whi = wi >> 2, wwi = wi & 3;
    int tr = token / WIN, tc = token % WIN;
    int hh = whi * WIN + tr, ww = wwi * WIN + tc;
    v = (_Float16)x[(((size_t)b * DMODEL + d) * 100 + hh) * 100 + ww];
  }
  xw[idx] = v;
}

// ---------------- fp32 -> f16 copy (weights)
__global__ __launch_bounds__(256) void k_cvt(const float* __restrict__ s, _Float16* __restrict__ d, int n) {
  int i = blockIdx.x * 256 + threadIdx.x;
  if (i < n) d[i] = (_Float16)s[i];
}

// ---------------- WMMA GEMM: C(MxN) = A(MxK f16 row-major) * B(NxK f16 row-major)^T
// Tile 128x64, 8 waves 4x2, wave tile 32x32 (4 wmma/K-step).
// Double-buffered LDS; staging via async global->LDS copies when available
// (fallback: register preload). Single barrier per K-step, peeled last step.
// EPI 0: f32 store | EPI 2: +bias,GELU,f16 | EPI 3: qkv split f16 | EPI 4: +bias,+residual,NCHW scatter
template <int EPI>
__global__ __launch_bounds__(256) void k_gemm(const _Float16* __restrict__ A, int lda,
                                              const _Float16* __restrict__ B, int ldb,
                                              int K, int ldc, const float* __restrict__ bias,
                                              void* __restrict__ o0v, void* __restrict__ o1v,
                                              void* __restrict__ o2v, const float* __restrict__ xres) {
  __shared__ _Float16 As[2][128 * 32];
  __shared__ _Float16 Bs[2][64 * 32];
  const int tid = threadIdx.x;
  const int lane = tid & 31, wave = tid >> 5;
  const int wm = wave >> 1, wn = wave & 1;
  const int bm = blockIdx.x * 128, bn = blockIdx.y * 64;
  const int mi = lane & 15, half = lane >> 4;
  const int arow = tid >> 1, aseg = (tid & 1) * 16;
  const int brow = tid >> 2, bseg = (tid & 3) * 8;
  const _Float16* aptr = A + (size_t)(bm + arow) * lda + aseg;
  const _Float16* bptr = B + (size_t)(bn + brow) * ldb + bseg;

  v8f c00 = {}, c01 = {}, c10 = {}, c11 = {};
  int cur = 0;

#if HAVE_ASYNC128
  auto stage = [&](int buf, int k0) {
    async_b128(aptr + k0,     &As[buf][arow * 32 + aseg]);
    async_b128(aptr + k0 + 8, &As[buf][arow * 32 + aseg + 8]);
    async_b128(bptr + k0,     &Bs[buf][brow * 32 + bseg]);
  };
  stage(0, 0);
  wait_async();
  __syncthreads();
#else
  *(v16h*)&As[0][arow * 32 + aseg] = *(const v16h*)aptr;
  *(v8h*)&Bs[0][brow * 32 + bseg]  = *(const v8h*)bptr;
  __syncthreads();
#endif

  auto step = [&](int buf) {
    v8h a0lo = *(const v8h*)&As[buf][(wm * 32 + mi) * 32 + half * 16];
    v8h a0hi = *(const v8h*)&As[buf][(wm * 32 + mi) * 32 + half * 16 + 8];
    v8h a1lo = *(const v8h*)&As[buf][(wm * 32 + 16 + mi) * 32 + half * 16];
    v8h a1hi = *(const v8h*)&As[buf][(wm * 32 + 16 + mi) * 32 + half * 16 + 8];
    v16h a0 = __builtin_shufflevector(a0lo, a0hi, 0,1,2,3,4,5,6,7,8,9,10,11,12,13,14,15);
    v16h a1 = __builtin_shufflevector(a1lo, a1hi, 0,1,2,3,4,5,6,7,8,9,10,11,12,13,14,15);
    v16h b0 = *(const v16h*)&Bs[buf][(wn * 32 + mi) * 32 + half * 16];
    v16h b1 = *(const v16h*)&Bs[buf][(wn * 32 + 16 + mi) * 32 + half * 16];
    c00 = wmma_f16(a0, b0, c00);
    c01 = wmma_f16(a0, b1, c01);
    c10 = wmma_f16(a1, b0, c10);
    c11 = wmma_f16(a1, b1, c11);
  };
  for (int k0 = 0; k0 + 32 < K; k0 += 32) {      // steady state
#if HAVE_ASYNC128
    stage(cur ^ 1, k0 + 32);                     // async copy overlaps the wmma below
    step(cur);
    wait_async();
    __syncthreads();
#else
    v16h aReg = *(const v16h*)(aptr + k0 + 32);
    v8h  bReg = *(const v8h*)(bptr + k0 + 32);
    step(cur);
    *(v16h*)&As[cur ^ 1][arow * 32 + aseg] = aReg;
    *(v8h*)&Bs[cur ^ 1][brow * 32 + bseg]  = bReg;
    __syncthreads();
#endif
    cur ^= 1;
  }
  step(cur);                                      // peeled last K-step

  const int r0 = bm + wm * 32;
  const int cA = bn + wn * 32 + mi;
  auto epi = [&](v8f cf, int rbase, int cbase) {
    const int rr0 = rbase + 8 * half;
    if constexpr (EPI == 0) {
      float* C = (float*)o0v;
#pragma unroll
      for (int r = 0; r < 8; ++r) C[(size_t)(rr0 + r) * ldc + cbase] = cf[r];
    } else if constexpr (EPI == 2) {
      _Float16* Ch = (_Float16*)o0v;
      const float bv = bias[cbase];
#pragma unroll
      for (int r = 0; r < 8; ++r) {
        float xv = cf[r] + bv;
        xv = 0.5f * xv * (1.f + erff(xv * 0.70710678118f));
        Ch[(size_t)(rr0 + r) * ldc + cbase] = (_Float16)xv;
      }
    } else if constexpr (EPI == 3) {
      const int part = cbase >> 8, cc = cbase & 255;
      const int head = cc >> 5, dhi = cc & 31;
      if (part == 0) {
        _Float16* qp = (_Float16*)o0v;
#pragma unroll
        for (int r = 0; r < 8; ++r)
          qp[((size_t)(rr0 + r) * NHEADS + head) * DHEAD + dhi] = (_Float16)(cf[r] * QSCALE);
      } else if (part == 1) {
        _Float16* kp = (_Float16*)o1v;
#pragma unroll
        for (int r = 0; r < 8; ++r)
          kp[((size_t)(rr0 + r) * NHEADS + head) * DHEAD + dhi] = (_Float16)cf[r];
      } else {
        _Float16* vp = (_Float16*)o2v;
        int wwin = rr0 / NPAD, token = rr0 - wwin * NPAD;   // one divide per fragment
#pragma unroll
        for (int r = 0; r < 8; ++r) {
          vp[(((size_t)wwin * NHEADS + head) * DHEAD + dhi) * NPAD + token] = (_Float16)cf[r];
          if (++token == NPAD) { token = 0; ++wwin; }
        }
      }
    } else {  // EPI == 4: bias + residual + NCHW scatter, incremental decomposition
      float* op = (float*)o0v;
      const float bv = bias[cbase];
      int wwin = rr0 / NPAD, token = rr0 - wwin * NPAD;
      int tr = token / WIN, tc = token - tr * WIN;
#pragma unroll
      for (int r = 0; r < 8; ++r) {
        if (token < NTOK) {
          const int b = wwin >> 4, wi = wwin & 15;
          const int hh = (wi >> 2) * WIN + tr;
          const int ww = (wi & 3) * WIN + tc;
          const size_t oidx = (((size_t)b * DMODEL + cbase) * 100 + hh) * 100 + ww;
          op[oidx] = xres[oidx] + (cf[r] + bv);
        }
        ++token; ++tc;
        if (tc == WIN) { tc = 0; ++tr; }
        if (token == NPAD) { token = 0; ++wwin; tr = 0; tc = 0; }
      }
    }
  };
  epi(c00, r0,      cA);
  epi(c01, r0,      cA + 16);
  epi(c10, r0 + 16, cA);
  epi(c11, r0 + 16, cA + 16);
}

// ---------------- flash attention per (window, head): 256 blocks, 8 waves, 5 lockstep passes.
// K/V tiles double-buffered in LDS (tiles depend only on jp), one barrier per jp step.
// Staging via async global->LDS b64 copies when available.
__global__ __launch_bounds__(256) void k_attn(const _Float16* __restrict__ q,
                                              const _Float16* __restrict__ k,
                                              const _Float16* __restrict__ vt,
                                              const float* __restrict__ bias_table,
                                              _Float16* __restrict__ out) {
  __shared__ float lbias[49 * 49];                  // 2401 per-head bias entries
  __shared__ _Float16 plds[8][16 * 32];             // per-wave P bounce buffer
  __shared__ _Float16 Ks[2][32 * 32];               // 32 tokens x 32 dh
  __shared__ _Float16 Vs[2][32 * 32];               // 32 dh x 32 tokens (dh-major)
  const int win = blockIdx.x >> 3, head = blockIdx.x & 7;
  const int tid = threadIdx.x, lane = tid & 31, wave = tid >> 5;
  for (int i = tid; i < 2401; i += 256) lbias[i] = bias_table[(size_t)i * NHEADS + head];
  const int mi = lane & 15, half = lane >> 4;
  const size_t vtbase = ((size_t)win * NHEADS + head) * DHEAD * NPAD;
  const int srow = tid >> 3;            // 0..31  (stage row)
  const int sseg = (tid & 7) * 4;       // 4 halves each
  const _Float16* kstage = k + ((size_t)(win * NPAD + srow) * NHEADS + head) * DHEAD + sseg;
  const _Float16* vstage = vt + vtbase + (size_t)srow * NPAD + sseg;

  v16h onesv;
#pragma unroll
  for (int i = 0; i < 16; ++i) onesv[i] = (_Float16)1.0f;

#if HAVE_ASYNC64
  auto stage = [&](int buf, int jp) {
    async_b64(kstage + (size_t)jp * 32 * NHEADS * DHEAD, &Ks[buf][srow * 32 + sseg]);
    async_b64(vstage + (size_t)jp * 32,                  &Vs[buf][srow * 32 + sseg]);
  };
  stage(0, 0);
  wait_async();
  __syncthreads();
#else
  {
    v4h kReg = *(const v4h*)kstage;
    v4h vReg = *(const v4h*)vstage;
    *(v4h*)&Ks[0][srow * 32 + sseg] = kReg;
    *(v4h*)&Vs[0][srow * 32 + sseg] = vReg;
  }
  __syncthreads();
#endif
  int cur = 0;

  for (int pass = 0; pass < 5; ++pass) {
    const int it = pass * 8 + wave;
    const int i0 = it * 16;
    // ---- Q A-fragment (16 x dh=32)
    const _Float16* qp = q + (((size_t)win * NPAD + (i0 + mi)) * NHEADS + head) * DHEAD;
    v8h qlo = *(const v8h*)(qp + half * 16);
    v8h qhi = *(const v8h*)(qp + half * 16 + 8);
    v16h qa = __builtin_shufflevector(qlo, qhi, 0,1,2,3,4,5,6,7,8,9,10,11,12,13,14,15);
    int qr[8], qc[8]; bool qv[8];
#pragma unroll
    for (int r = 0; r < 8; ++r) {
      int qi = i0 + r + 8 * half;
      qv[r] = qi < NTOK;
      qr[r] = qi / WIN; qc[r] = qi % WIN;
    }
    float lrow[8];
    float mcur = -1e30f;                       // tile-wide running max (valid per-row offset)
    v8f o0 = {}, o1 = {};
#pragma unroll
    for (int r = 0; r < 8; ++r) lrow[r] = 0.f;

    for (int jp = 0; jp < NPAD / 32; ++jp) {   // 20 iterations, 32 keys each
      // next tile index (next jp, or jp=0 of next pass — same data sequence)
      const int nj = (jp + 1 == NPAD / 32) ? 0 : jp + 1;
#if HAVE_ASYNC64
      stage(cur ^ 1, nj);                      // async copy overlaps all compute below
#else
      v4h kReg = *(const v4h*)(kstage + (size_t)nj * 32 * NHEADS * DHEAD);
      v4h vReg = *(const v4h*)(vstage + (size_t)nj * 32);
#endif

      v16h kb0 = *(const v16h*)&Ks[cur][mi * 32 + half * 16];
      v16h kb1 = *(const v16h*)&Ks[cur][(16 + mi) * 32 + half * 16];
      v8f s0 = {}, s1 = {};
      s0 = wmma_f16(qa, kb0, s0);
      s1 = wmma_f16(qa, kb1, s1);
      // ---- relative-position bias + masking (branchless: clamped index, cndmask select)
      const int kt0 = jp * 32 + mi, kt1 = kt0 + 16;
      const int kr0 = kt0 / WIN, kc0 = kt0 % WIN;
      const int kr1 = kt1 / WIN, kc1 = kt1 % WIN;
      const bool kv0 = kt0 < NTOK, kv1 = kt1 < NTOK;
#pragma unroll
      for (int r = 0; r < 8; ++r) {
        int ix0 = (qr[r] - kr0 + 24) * 49 + (qc[r] - kc0 + 24);
        int ix1 = (qr[r] - kr1 + 24) * 49 + (qc[r] - kc1 + 24);
        ix0 = min(max(ix0, 0), 2400);
        ix1 = min(max(ix1, 0), 2400);
        float bv0 = lbias[ix0];
        float bv1 = lbias[ix1];
        s0[r] = (qv[r] && kv0) ? (s0[r] + bv0) : -1e30f;
        s1[r] = (qv[r] && kv1) ? (s1[r] + bv1) : -1e30f;
      }
      // ---- tile-wide max: lane-local tree + 5 butterfly stages over all 32 lanes
      float tmax = fmaxf(s0[0], s1[0]);
#pragma unroll
      for (int r = 1; r < 8; ++r) tmax = fmaxf(tmax, fmaxf(s0[r], s1[r]));
      tmax = fmaxf(tmax, __shfl_xor(tmax, 1));
      tmax = fmaxf(tmax, __shfl_xor(tmax, 2));
      tmax = fmaxf(tmax, __shfl_xor(tmax, 4));
      tmax = fmaxf(tmax, __shfl_xor(tmax, 8));
      tmax = fmaxf(tmax, __shfl_xor(tmax, 16));
      float mnew = fmaxf(mcur, tmax);
      float sc = __expf(mcur - mnew);
      mcur = mnew;
#pragma unroll
      for (int r = 0; r < 8; ++r) {
        s0[r] = __expf(s0[r] - mnew);
        s1[r] = __expf(s1[r] - mnew);
        o0[r] *= sc; o1[r] *= sc;
        lrow[r] *= sc;
      }
      // ---- bounce P through LDS: C-layout -> A-layout (wave-synchronous)
#pragma unroll
      for (int r = 0; r < 8; ++r) {
        plds[wave][(r + 8 * half) * 32 + mi]      = (_Float16)s0[r];
        plds[wave][(r + 8 * half) * 32 + 16 + mi] = (_Float16)s1[r];
      }
      asm volatile("s_wait_dscnt 0" ::: "memory");
      v8h plo = *(const v8h*)&plds[wave][mi * 32 + half * 16];
      v8h phi = *(const v8h*)&plds[wave][mi * 32 + half * 16 + 8];
      v16h pa = __builtin_shufflevector(plo, phi, 0,1,2,3,4,5,6,7,8,9,10,11,12,13,14,15);
      // ---- row-sums via WMMA against all-ones B: lands exactly on lrow slots
      v8f zz = {};
      v8f rsum = wmma_f16(pa, onesv, zz);
#pragma unroll
      for (int r = 0; r < 8; ++r) lrow[r] += rsum[r];
      // ---- V B-fragments from LDS (dh-major -> contiguous 32B)
      v16h vb0 = *(const v16h*)&Vs[cur][mi * 32 + half * 16];
      v16h vb1 = *(const v16h*)&Vs[cur][(16 + mi) * 32 + half * 16];
      o0 = wmma_f16(pa, vb0, o0);
      o1 = wmma_f16(pa, vb1, o1);
      // ---- publish next tile, single barrier
#if HAVE_ASYNC64
      wait_async();
#else
      *(v4h*)&Ks[cur ^ 1][srow * 32 + sseg] = kReg;
      *(v4h*)&Vs[cur ^ 1][srow * 32 + sseg] = vReg;
#endif
      __syncthreads();
      cur ^= 1;
    }
    // ---- normalize (fast rcp) + store (head-major concat layout == (win,tok,h,dh))
#pragma unroll
    for (int r = 0; r < 8; ++r) {
      int qi = i0 + r + 8 * half;
      float inv = qv[r] ? __builtin_amdgcn_rcpf(lrow[r]) : 0.f;
      _Float16* op = out + (((size_t)win * NPAD + qi) * NHEADS + head) * DHEAD;
      op[mi]      = (_Float16)(o0[r] * inv);
      op[16 + mi] = (_Float16)(o1[r] * inv);
    }
  }
}

// ---------------- LayerNorm over d=256, one wave per row (no LDS, shuffle-only)
__global__ __launch_bounds__(256) void k_ln(const float* __restrict__ xin,
                                            const float* __restrict__ g,
                                            const float* __restrict__ b,
                                            _Float16* __restrict__ y) {
  const int wave = threadIdx.x >> 5, lane = threadIdx.x & 31;
  const int row = blockIdx.x * 8 + wave;
  const float* rp = xin + (size_t)row * DMODEL + lane * 8;
  float4 lo = *(const float4*)rp;
  float4 hi = *(const float4*)(rp + 4);
  float v[8] = {lo.x, lo.y, lo.z, lo.w, hi.x, hi.y, hi.z, hi.w};
  float s = 0.f;
#pragma unroll
  for (int i = 0; i < 8; ++i) s += v[i];
  s += __shfl_xor(s, 1); s += __shfl_xor(s, 2); s += __shfl_xor(s, 4);
  s += __shfl_xor(s, 8); s += __shfl_xor(s, 16);
  const float mean = s * (1.f / 256.f);
  float s2 = 0.f;
#pragma unroll
  for (int i = 0; i < 8; ++i) { v[i] -= mean; s2 += v[i] * v[i]; }
  s2 += __shfl_xor(s2, 1); s2 += __shfl_xor(s2, 2); s2 += __shfl_xor(s2, 4);
  s2 += __shfl_xor(s2, 8); s2 += __shfl_xor(s2, 16);
  const float rstd = rsqrtf(s2 * (1.f / 256.f) + 1e-5f);
  float4 glo = *(const float4*)(g + lane * 8);
  float4 ghi = *(const float4*)(g + lane * 8 + 4);
  float4 blo = *(const float4*)(b + lane * 8);
  float4 bhi = *(const float4*)(b + lane * 8 + 4);
  float gg[8] = {glo.x, glo.y, glo.z, glo.w, ghi.x, ghi.y, ghi.z, ghi.w};
  float bb[8] = {blo.x, blo.y, blo.z, blo.w, bhi.x, bhi.y, bhi.z, bhi.w};
  v8h ov;
#pragma unroll
  for (int i = 0; i < 8; ++i) ov[i] = (_Float16)(v[i] * rstd * gg[i] + bb[i]);
  *(v8h*)(y + (size_t)row * DMODEL + lane * 8) = ov;
}

extern "C" void kernel_launch(void* const* d_in, const int* in_sizes, int n_in,
                              void* d_out, int out_size, void* d_ws, size_t ws_size,
                              hipStream_t stream) {
  const float* x          = (const float*)d_in[0];
  const float* w_qkv      = (const float*)d_in[1];
  const float* w_out      = (const float*)d_in[2];
  const float* bias_table = (const float*)d_in[3];
  const float* gamma2     = (const float*)d_in[4];
  const float* beta2      = (const float*)d_in[5];
  const float* w1         = (const float*)d_in[6];
  const float* b1         = (const float*)d_in[7];
  const float* w2         = (const float*)d_in[8];
  const float* b2         = (const float*)d_in[9];
  float* out = (float*)d_out;

  size_t off = 0;
  auto take = [&](size_t bytes) -> void* {
    void* p = (char*)d_ws + off;
    off += (bytes + 255) & ~(size_t)255;
    return p;
  };
  _Float16* xw    = (_Float16*)take((size_t)MROWS * DMODEL * 2);
  _Float16* wqkvh = (_Float16*)take((size_t)768 * 256 * 2);
  _Float16* wouth = (_Float16*)take((size_t)256 * 256 * 2);
  _Float16* w1h   = (_Float16*)take((size_t)512 * 256 * 2);
  _Float16* w2h   = (_Float16*)take((size_t)256 * 512 * 2);
  _Float16* qh    = (_Float16*)take((size_t)MROWS * 256 * 2);
  _Float16* kh    = (_Float16*)take((size_t)MROWS * 256 * 2);
  _Float16* vth   = (_Float16*)take((size_t)MROWS * 256 * 2);
  _Float16* attn  = (_Float16*)take((size_t)MROWS * 256 * 2);
  float*    projf = (float*)take((size_t)MROWS * 256 * 4);
  _Float16* yh    = (_Float16*)take((size_t)MROWS * 256 * 2);
  _Float16* hidh  = (_Float16*)take((size_t)MROWS * 512 * 2);

  k_winpart<<<(MROWS * DMODEL) / 256, 256, 0, stream>>>(x, xw);
  k_cvt<<<(768 * 256 + 255) / 256, 256, 0, stream>>>(w_qkv, wqkvh, 768 * 256);
  k_cvt<<<(256 * 256 + 255) / 256, 256, 0, stream>>>(w_out, wouth, 256 * 256);
  k_cvt<<<(512 * 256 + 255) / 256, 256, 0, stream>>>(w1, w1h, 512 * 256);
  k_cvt<<<(256 * 512 + 255) / 256, 256, 0, stream>>>(w2, w2h, 256 * 512);

  // QKV projection, split epilogue -> q/k/vt f16
  k_gemm<3><<<dim3(MROWS / 128, 768 / 64), 256, 0, stream>>>(
      xw, 256, wqkvh, 256, 256, 768, nullptr, qh, kh, vth, nullptr);
  // windowed attention
  k_attn<<<NWIN * NHEADS, 256, 0, stream>>>(qh, kh, vth, bias_table, attn);
  // out projection -> f32
  k_gemm<0><<<dim3(MROWS / 128, 256 / 64), 256, 0, stream>>>(
      attn, 256, wouth, 256, 256, 256, nullptr, projf, nullptr, nullptr, nullptr);
  // layernorm -> f16 (wave per row)
  k_ln<<<MROWS / 8, 256, 0, stream>>>(projf, gamma2, beta2, yh);
  // MLP1 + bias + GELU -> f16
  k_gemm<2><<<dim3(MROWS / 128, 512 / 64), 256, 0, stream>>>(
      yh, 256, w1h, 256, 256, 512, b1, hidh, nullptr, nullptr, nullptr);
  // MLP2 + bias + residual + NCHW scatter -> out
  k_gemm<4><<<dim3(MROWS / 128, 256 / 64), 256, 0, stream>>>(
      hidh, 512, w2h, 512, 512, 256, b2, out, nullptr, nullptr, x);
}
